// MoEModel_3762391351644
// MI455X (gfx1250) — compile-verified, matrix operand order checked by reading, not run on hardware
//
#include <hip/hip_runtime.h>

// ---------------- WMMA types / helpers (CDNA5 gfx1250, wave32) ----------------
typedef __attribute__((ext_vector_type(16))) __bf16 v16bf;
typedef __attribute__((ext_vector_type(8)))  float  v8f;

union Frag {
  v16bf v;
  unsigned int   u[8];
  unsigned short s[16];
};

__device__ __forceinline__ unsigned short f2bf(float f) {
  __bf16 h = (__bf16)f;
  unsigned short r;
  __builtin_memcpy(&r, &h, 2);
  return r;
}

// 16-bit A/B matrix lane layout (ISA 7.12.2): lanes 0-15 hold K {0..7,16..23},
// lanes 16-31 hold K {8..15,24..31}; VGPR v holds K pair (kbase, kbase+1).
__device__ __forceinline__ int kbase(int v, int half) {
  return (v < 4 ? 2 * v : 16 + 2 * (v - 4)) + 8 * half;
}

// Generic (LDS-backed) pointer -> 32-bit LDS byte offset for async-to-LDS ops.
__device__ __forceinline__ unsigned lds_off(const void* p) {
  return (unsigned)(unsigned long long)
      (__attribute__((address_space(3))) const void*)p;
}

// ASYNCcnt-tracked bulk copy: 16B global -> LDS without VGPR round trip
// (GLOBAL_LOAD_ASYNC_TO_LDS_B128, CDNA5 ISA opcode 98).
__device__ __forceinline__ void async_g2l_b128(unsigned lds_addr,
                                               const void* gaddr) {
  asm volatile("global_load_async_to_lds_b128 %0, %1, off"
               :: "v"(lds_addr), "v"(gaddr)
               : "memory");
}

__device__ __forceinline__ void wait_async0() {
  asm volatile("s_wait_asynccnt 0x0" ::: "memory");
}

// ---------------- Problem constants ----------------
#define BATCH 4096
#define NEXP  4
#define NCLS  10

// ---------------- Workspace layout (bytes) ----------------
#define OFF_POOLED 0u               // 4096*16 f32           = 262144
#define OFF_CNT    262144u          // 4 int
#define OFF_ASSIGN 262400u          // 4096 int              = 16384
#define OFF_WVAL   278784u          // 4096 f32              = 16384
#define OFF_LISTS  295168u          // 4*4096 int            = 65536
#define OFF_WB1    360704u          // 4*32*32 bf16          = 8192
#define OFF_WB2    368896u          // 4*64*288 bf16         = 147456
#define OFF_WFC2   516352u          // 4*16*128 bf16         = 16384
#define OFF_WFC1   532736u          // 4*128*4096 bf16       = 4194304
#define OFF_FEATS  4727040u         // 4096*4096 bf16        = 33554432

// =====================================================================
// Kernel 1: weight repack fp32 -> bf16, [n][k] rows (k-pairs contiguous)
// =====================================================================
__global__ void prep_kernel(const float* __restrict__ ew1,   // (E,32,3,3,3)
                            const float* __restrict__ ew2,   // (E,64,32,3,3)
                            const float* __restrict__ ef1,   // (E,128,4096)
                            const float* __restrict__ ef2,   // (E,10,128)
                            unsigned short* __restrict__ wb1,
                            unsigned short* __restrict__ wb2,
                            unsigned short* __restrict__ wfc1,
                            unsigned short* __restrict__ wfc2,
                            int* __restrict__ cnt) {
  int id = blockIdx.x * blockDim.x + threadIdx.x;
  if (id < NEXP) cnt[id] = 0;
  if (id >= 2183168) return;
  if (id < 4096) {                       // conv1: K 27 -> pad 32
    int k = id & 31, en = id >> 5;
    wb1[id] = (k < 27) ? f2bf(ew1[en * 27 + k]) : (unsigned short)0;
  } else if (id < 77824) {               // conv2: direct (K=288)
    int j = id - 4096;
    wb2[j] = f2bf(ew2[j]);
  } else if (id < 86016) {               // fc2: N 10 -> pad 16
    int j = id - 77824;
    int e = j >> 11, rem = j & 2047, n = rem >> 7, k = rem & 127;
    wfc2[j] = (n < NCLS) ? f2bf(ef2[(e * NCLS + n) * 128 + k]) : (unsigned short)0;
  } else {                               // fc1: direct
    int j = id - 86016;
    wfc1[j] = f2bf(ef1[j]);
  }
}

// =====================================================================
// Kernel 2: router conv(3->16) + relu + global avg pool  (fp32 VALU)
// =====================================================================
__global__ void router_conv_kernel(const float* __restrict__ x,
                                   const float* __restrict__ gw,
                                   const float* __restrict__ gb,
                                   float* __restrict__ pooled) {
  __shared__ float xr[3 * 34 * 34];
  __shared__ float wsm[16 * 27];
  __shared__ float part[256];
  const int tid = threadIdx.x;
  const int img = blockIdx.x;
  for (int j = tid; j < 3468; j += 256) xr[j] = 0.f;
  for (int j = tid; j < 432;  j += 256) wsm[j] = gw[j];
  __syncthreads();
  const float* xg = x + (size_t)img * 3072;
  for (int j = tid; j < 3072; j += 256) {
    int ci = j >> 10, rem = j & 1023, y = rem >> 5, xx = rem & 31;
    xr[ci * 1156 + (y + 1) * 34 + (xx + 1)] = xg[j];
  }
  __syncthreads();
  const int c = tid >> 4, s = tid & 15;
  const float bias = gb[c];
  float loc = 0.f;
  for (int p = 0; p < 64; ++p) {
    int m = s * 64 + p;
    int y = m >> 5, xx = m & 31;
    float a = bias;
#pragma unroll
    for (int ci = 0; ci < 3; ++ci)
#pragma unroll
      for (int rr = 0; rr < 3; ++rr)
#pragma unroll
        for (int dd = 0; dd < 3; ++dd)
          a += wsm[c * 27 + ci * 9 + rr * 3 + dd] *
               xr[ci * 1156 + (y + rr) * 34 + (xx + dd)];
    loc += fmaxf(a, 0.f);
  }
  part[tid] = loc;
  __syncthreads();
  if (s == 0) {                      // deterministic per-channel sum
    float t = 0.f;
    for (int q = 0; q < 16; ++q) t += part[c * 16 + q];
    pooled[img * 16 + c] = t * (1.f / 1024.f);
  }
}

// =====================================================================
// Kernel 3: router FC + softmax + top-1, build expert compaction lists
// =====================================================================
__global__ void router_fc_kernel(const float* __restrict__ pooled,
                                 const float* __restrict__ gwfc,
                                 const float* __restrict__ gbfc,
                                 float* __restrict__ probs,
                                 int* __restrict__ assign,
                                 float* __restrict__ wvalp,
                                 int* __restrict__ cnt,
                                 int* __restrict__ lists) {
  int i = blockIdx.x * blockDim.x + threadIdx.x;
  if (i >= BATCH) return;
  float p[16];
#pragma unroll
  for (int j = 0; j < 16; ++j) p[j] = pooled[i * 16 + j];
  float lg[NEXP];
  float mx = -1e30f;
#pragma unroll
  for (int e = 0; e < NEXP; ++e) {
    float s = gbfc[e];
#pragma unroll
    for (int j = 0; j < 16; ++j) s += gwfc[e * 16 + j] * p[j];
    lg[e] = s;
    mx = fmaxf(mx, s);
  }
  float den = 0.f;
#pragma unroll
  for (int e = 0; e < NEXP; ++e) { lg[e] = expf(lg[e] - mx); den += lg[e]; }
  float inv = 1.f / den;
  int best = 0;
  float bp = -1.f;
#pragma unroll
  for (int e = 0; e < NEXP; ++e) {
    float pr = lg[e] * inv;
    probs[i * NEXP + e] = pr;
    if (pr > bp) { bp = pr; best = e; }   // strict > == first-max tie rule
  }
  assign[i] = best;
  wvalp[i] = bp;
  int slot = atomicAdd(&cnt[best], 1);    // order-independent result
  lists[best * BATCH + slot] = i;
}

// =====================================================================
// Kernel 4: aux load-balance loss (deterministic tree reduction)
// =====================================================================
__global__ void aux_kernel(const float* __restrict__ probs,
                           float* __restrict__ aux) {
  __shared__ float red[256 * NEXP];
  int tid = threadIdx.x;
  float s[NEXP] = {0.f, 0.f, 0.f, 0.f};
  for (int i = tid; i < BATCH; i += 256)
#pragma unroll
    for (int e = 0; e < NEXP; ++e) s[e] += probs[i * NEXP + e];
#pragma unroll
  for (int e = 0; e < NEXP; ++e) red[tid * NEXP + e] = s[e];
  __syncthreads();
  for (int st = 128; st > 0; st >>= 1) {
    if (tid < st)
#pragma unroll
      for (int e = 0; e < NEXP; ++e)
        red[tid * NEXP + e] += red[(tid + st) * NEXP + e];
    __syncthreads();
  }
  if (tid == 0) {
    float a = 0.f;
#pragma unroll
    for (int e = 0; e < NEXP; ++e) {
      float d = red[e] * (1.f / (float)BATCH) - 1.f / (float)NEXP;
      a += d * d;
    }
    aux[0] = a * (1.f / (float)NEXP);
  }
}

// =====================================================================
// Kernel 5: expert convs (implicit-GEMM WMMA bf16), top-1 gathered.
// One workgroup (8 waves) per image. Dynamic LDS = 115744 B.
//   weights staged via GLOBAL_LOAD_ASYNC_TO_LDS_B128 (ASYNCcnt)
//   conv1: M=1024 (32x32), K=32(27), N=32   -> relu+maxpool via atomicMax
//   conv2: M=256 (16x16),  K=288,    N=64   -> relu+maxpool via atomicMax
//   features (64*8*8) -> bf16 to global ws
// =====================================================================
__global__ void expert_conv_kernel(const float* __restrict__ x,
                                   const int* __restrict__ assign,
                                   const unsigned short* __restrict__ wb1g,
                                   const unsigned short* __restrict__ wb2g,
                                   const float* __restrict__ eb1,
                                   const float* __restrict__ eb2,
                                   unsigned short* __restrict__ feats) {
  extern __shared__ char smem[];
  unsigned short* xpad = (unsigned short*)smem;             // [3][34][34] bf16
  unsigned short* wb1s = (unsigned short*)(smem + 6944);    // [32n][32k] bf16
  float*          p1   = (float*)(smem + 8992);             // [32][16][16] f32
  unsigned short* p1bf = (unsigned short*)(smem + 41760);   // [32][18][18] bf16
  unsigned short* wb2s = (unsigned short*)(smem + 62496);   // [64n][288k] bf16
  float*          p2   = (float*)(smem + 99360);            // [64][8][8] f32
  const int tid  = threadIdx.x;
  const int img  = blockIdx.x;
  const int e    = assign[img];
  const int lane = tid & 31;
  const int wv   = tid >> 5;
  const int half = lane >> 4;
  const float* xg = x + (size_t)img * 3072;
  __builtin_prefetch(xg, 0, 0);           // global_prefetch_b8 toward WGP

  // ---- phase 0a: async-stage conv weights into LDS (no VGPR round trip) ----
  {
    const unsigned wb1_l = lds_off(wb1s);
    const char* wb1_gp = (const char*)(wb1g + e * 1024);
    if (tid < 128)
      async_g2l_b128(wb1_l + (unsigned)tid * 16u, wb1_gp + tid * 16);
    const unsigned wb2_l = lds_off(wb2s);
    const char* wb2_gp = (const char*)(wb2g + e * 64 * 288);
    for (int j = tid; j < 2304; j += 256)           // 36864 B
      async_g2l_b128(wb2_l + (unsigned)j * 16u, wb2_gp + j * 16);
  }

  // ---- phase 0b: zero accumulators while the async copies fly ----
  unsigned int* xpadU = (unsigned int*)xpad;
  for (int j = tid; j < 1734; j += 256) xpadU[j] = 0u;
  for (int j = tid; j < 8192; j += 256) p1[j] = 0.f;     // init 0 => fused relu
  unsigned int* p1bfU = (unsigned int*)p1bf;
  for (int j = tid; j < 5184; j += 256) p1bfU[j] = 0u;
  for (int j = tid; j < 4096; j += 256) p2[j] = 0.f;
  wait_async0();
  __syncthreads();

  // ---- phase 1: load image, bf16, zero-padded halo ----
  for (int j = tid; j < 3072; j += 256) {
    int ci = j >> 10, rem = j & 1023, y = rem >> 5, xx = rem & 31;
    xpad[ci * 1156 + (y + 1) * 34 + (xx + 1)] = f2bf(xg[j]);
  }
  __syncthreads();

  const unsigned int* wb1su = (const unsigned int*)wb1s;
  const unsigned int* wb2su = (const unsigned int*)wb2s;

  // ---- phase 2: conv1 WMMA (64 M-tiles, 2 N-halves, single K step) ----
  int* p1i = (int*)p1;
  for (int t = wv * 8; t < wv * 8 + 8; ++t) {
    const int m0 = t << 4;
    Frag a;
    {
      int m = m0 + (lane & 15);
      int y = m >> 5, xx = m & 31;
#pragma unroll
      for (int v = 0; v < 8; ++v) {
        int kb = kbase(v, half);
#pragma unroll
        for (int q = 0; q < 2; ++q) {
          int k = kb + q;
          unsigned short val = 0;
          if (k < 27) {
            int ci = k / 9, rr = (k % 9) / 3, dd = k % 3;
            val = xpad[ci * 1156 + (y + rr) * 34 + (xx + dd)];
          }
          a.s[2 * v + q] = val;
        }
      }
    }
#pragma unroll
    for (int h2 = 0; h2 < 2; ++h2) {
      Frag b;
      int n = (lane & 15) + 16 * h2;
#pragma unroll
      for (int v = 0; v < 8; ++v)
        b.u[v] = wb1su[n * 16 + (kbase(v, half) >> 1)];
      v8f c;
#pragma unroll
      for (int r = 0; r < 8; ++r) c[r] = 0.f;
      c = __builtin_amdgcn_wmma_f32_16x16x32_bf16(false, a.v, false, b.v,
                                                  (short)0, c, false, false);
      float bias = eb1[e * 32 + n];
#pragma unroll
      for (int r = 0; r < 8; ++r) {
        int mm = m0 + r + 8 * half;
        int yy = mm >> 5, xc = mm & 31;
        float val = c[r] + bias;
        // max(0, vals) == relu+maxpool; int-compare valid vs 0-init
        atomicMax(&p1i[n * 256 + (yy >> 1) * 16 + (xc >> 1)],
                  __float_as_int(val));
      }
    }
  }
  __syncthreads();

  // ---- phase 3: pooled1 -> bf16 with halo padding ----
  for (int j = tid; j < 8192; j += 256) {
    int ci = j >> 8, rem = j & 255, y = rem >> 4, xx = rem & 15;
    p1bf[ci * 324 + (y + 1) * 18 + (xx + 1)] = f2bf(p1[j]);
  }
  __syncthreads();

  // ---- phase 4: conv2 WMMA (16 M-tiles, 4 N-tiles, 9 K steps) ----
  int* p2i = (int*)p2;
  for (int mt = wv * 2; mt < wv * 2 + 2; ++mt) {
    const int m0 = mt << 4;
    const int m = m0 + (lane & 15);
    const int y = m >> 4, xx = m & 15;
    v8f acc[4];
#pragma unroll
    for (int nt = 0; nt < 4; ++nt)
#pragma unroll
      for (int r = 0; r < 8; ++r) acc[nt][r] = 0.f;
    for (int kf = 0; kf < 9; ++kf) {
      Frag a;
#pragma unroll
      for (int v = 0; v < 8; ++v) {
        int kb = kf * 32 + kbase(v, half);
#pragma unroll
        for (int q = 0; q < 2; ++q) {
          int k = kb + q;
          int ci = k / 9, rr = (k % 9) / 3, dd = k % 3;
          a.s[2 * v + q] = p1bf[ci * 324 + (y + rr) * 18 + (xx + dd)];
        }
      }
#pragma unroll
      for (int nt = 0; nt < 4; ++nt) {
        Frag b;
        int n = nt * 16 + (lane & 15);
#pragma unroll
        for (int v = 0; v < 8; ++v)
          b.u[v] = wb2su[n * 144 + ((kf * 32 + kbase(v, half)) >> 1)];
        acc[nt] = __builtin_amdgcn_wmma_f32_16x16x32_bf16(
            false, a.v, false, b.v, (short)0, acc[nt], false, false);
      }
    }
#pragma unroll
    for (int nt = 0; nt < 4; ++nt) {
      int n = nt * 16 + (lane & 15);
      float bias = eb2[e * 64 + n];
#pragma unroll
      for (int r = 0; r < 8; ++r) {
        int mm = m0 + r + 8 * half;
        int yy = mm >> 4, xc = mm & 15;
        float val = acc[nt][r] + bias;
        atomicMax(&p2i[n * 64 + (yy >> 1) * 8 + (xc >> 1)],
                  __float_as_int(val));
      }
    }
  }
  __syncthreads();

  // ---- phase 5: features (c*64 + y*8 + x order) -> bf16 global ----
  unsigned int* featU = (unsigned int*)feats;
  size_t base = (size_t)img * 2048;   // u32 units
  int tb = tid * 16;
#pragma unroll
  for (int t = 0; t < 8; ++t) {
    unsigned int lo = f2bf(p2[tb + 2 * t]);
    unsigned int hi = f2bf(p2[tb + 2 * t + 1]);
    featU[base + (size_t)tid * 8 + t] = lo | (hi << 16);
  }
}

// =====================================================================
// Kernel 6: FC1 (16img x 4096 x 128) + FC2 (16img x 128 x 16) per tile,
// per-expert compacted lists, scaled scatter to final output.
// grid = E * (BATCH/16) tiles, 8 waves each.
// =====================================================================
__global__ void fc_kernel(const unsigned short* __restrict__ feats,
                          const int* __restrict__ lists,
                          const int* __restrict__ cnt,
                          const float* __restrict__ wvalp,
                          const unsigned short* __restrict__ wfc1,
                          const float* __restrict__ ebfc1,
                          const unsigned short* __restrict__ wfc2,
                          const float* __restrict__ ebfc2,
                          float* __restrict__ out) {
  __shared__ int   simg[16];
  __shared__ float swv[16];
  __shared__ unsigned short fcbuf[16 * 128];
  const int e = blockIdx.x >> 8;
  const int t = blockIdx.x & 255;
  const int c = cnt[e];
  const int base = t << 4;
  if (base >= c) return;           // uniform per block
  const int tid = threadIdx.x;
  if (tid < 16) {
    int j = base + tid;
    int img = lists[e * BATCH + (j < c ? j : base)];  // pad with dup (no store)
    simg[tid] = img;
    swv[tid] = wvalp[img];
  }
  __syncthreads();
  const int lane = tid & 31;
  const int wv = tid >> 5;
  const int half = lane >> 4;
  const int mloc = lane & 15;
  const unsigned int* featU = (const unsigned int*)feats;
  const unsigned int* w1U = (const unsigned int*)wfc1;

  // FC1: wave w -> outputs [w*16, w*16+16), K = 4096 = 128 WMMA steps
  const int n0 = wv * 16;
  v8f acc;
#pragma unroll
  for (int r = 0; r < 8; ++r) acc[r] = 0.f;
  const size_t arow = (size_t)simg[mloc] * 2048;              // u32 units
  const size_t brow = (size_t)(e * 128 + n0 + mloc) * 2048;   // u32 units
  for (int kf = 0; kf < 128; ++kf) {
    Frag a, b;
#pragma unroll
    for (int v = 0; v < 8; ++v) {
      int kb2 = (kf * 32 + kbase(v, half)) >> 1;
      a.u[v] = featU[arow + kb2];
      b.u[v] = w1U[brow + kb2];
    }
    acc = __builtin_amdgcn_wmma_f32_16x16x32_bf16(false, a.v, false, b.v,
                                                  (short)0, acc, false, false);
  }
  {
    int n = n0 + mloc;
    float bias = ebfc1[e * 128 + n];
#pragma unroll
    for (int r = 0; r < 8; ++r) {
      int m = r + 8 * half;
      float v = acc[r] + bias;
      fcbuf[m * 128 + n] = f2bf(v > 0.f ? v : 0.f);
    }
  }
  __syncthreads();

  // FC2 on wave 0: K=128 -> 4 WMMA steps, N=16 (10 valid)
  if (wv == 0) {
    const unsigned int* fcU = (const unsigned int*)fcbuf;
    const unsigned int* w2U = (const unsigned int*)wfc2;
    v8f acc2;
#pragma unroll
    for (int r = 0; r < 8; ++r) acc2[r] = 0.f;
    for (int kf = 0; kf < 4; ++kf) {
      Frag a, b;
#pragma unroll
      for (int v = 0; v < 8; ++v) {
        int kb = kf * 32 + kbase(v, half);
        a.u[v] = fcU[(mloc * 128 + kb) >> 1];
        b.u[v] = w2U[((e * 16 + mloc) * 128 + kb) >> 1];
      }
      acc2 = __builtin_amdgcn_wmma_f32_16x16x32_bf16(
          false, a.v, false, b.v, (short)0, acc2, false, false);
    }
    int n = mloc;
    if (n < NCLS) {
      float bias = ebfc2[e * NCLS + n];
#pragma unroll
      for (int r = 0; r < 8; ++r) {
        int m = r + 8 * half;
        if (base + m < c) {
          int img = simg[m];
          out[img * NCLS + n] = (acc2[r] + bias) * swv[m];
        }
      }
    }
  }
}

// =====================================================================
extern "C" void kernel_launch(void* const* d_in, const int* in_sizes, int n_in,
                              void* d_out, int out_size, void* d_ws,
                              size_t ws_size, hipStream_t stream) {
  const float* x       = (const float*)d_in[0];
  const float* gw_conv = (const float*)d_in[1];
  const float* gb_conv = (const float*)d_in[2];
  const float* gw_fc   = (const float*)d_in[3];
  const float* gb_fc   = (const float*)d_in[4];
  const float* ew1     = (const float*)d_in[5];
  const float* eb1     = (const float*)d_in[6];
  const float* ew2     = (const float*)d_in[7];
  const float* eb2     = (const float*)d_in[8];
  const float* ef1     = (const float*)d_in[9];
  const float* ebf1    = (const float*)d_in[10];
  const float* ef2     = (const float*)d_in[11];
  const float* ebf2    = (const float*)d_in[12];

  char* ws = (char*)d_ws;
  float* pooled          = (float*)(ws + OFF_POOLED);
  int*   cnt             = (int*)(ws + OFF_CNT);
  int*   assign          = (int*)(ws + OFF_ASSIGN);
  float* wvalp           = (float*)(ws + OFF_WVAL);
  int*   lists           = (int*)(ws + OFF_LISTS);
  unsigned short* wb1    = (unsigned short*)(ws + OFF_WB1);
  unsigned short* wb2    = (unsigned short*)(ws + OFF_WB2);
  unsigned short* wfc2   = (unsigned short*)(ws + OFF_WFC2);
  unsigned short* wfc1   = (unsigned short*)(ws + OFF_WFC1);
  unsigned short* feats  = (unsigned short*)(ws + OFF_FEATS);

  float* out_final = (float*)d_out;                 // (4096,10)
  float* out_probs = out_final + BATCH * NCLS;      // (4096,4)
  float* out_aux   = out_final + BATCH * NCLS + BATCH * NEXP;  // scalar

  (void)hipFuncSetAttribute((const void*)expert_conv_kernel,
                            hipFuncAttributeMaxDynamicSharedMemorySize, 115744);

  prep_kernel<<<8529, 256, 0, stream>>>(ew1, ew2, ef1, ef2, wb1, wb2, wfc1,
                                        wfc2, cnt);
  router_conv_kernel<<<BATCH, 256, 0, stream>>>(x, gw_conv, gb_conv, pooled);
  router_fc_kernel<<<BATCH / 256, 256, 0, stream>>>(pooled, gw_fc, gb_fc,
                                                    out_probs, assign, wvalp,
                                                    cnt, lists);
  aux_kernel<<<1, 256, 0, stream>>>(out_probs, out_aux);
  expert_conv_kernel<<<BATCH, 256, 115744, stream>>>(x, assign, wb1, wb2, eb1,
                                                     eb2, feats);
  fc_kernel<<<NEXP * (BATCH / 16), 256, 0, stream>>>(feats, lists, cnt, wvalp,
                                                     wfc1, ebf1, wfc2, ebf2,
                                                     out_final);
}
